// Discriminator_57028575756563
// MI455X (gfx1250) — compile-verified, matrix operand order checked by reading, not run on hardware
//
#include <hip/hip_runtime.h>

typedef __attribute__((ext_vector_type(16))) _Float16 v16h;
typedef __attribute__((ext_vector_type(8)))  _Float16 v8h;
typedef __attribute__((ext_vector_type(8)))  float    v8f;

#define WMMA(a,b,c) __builtin_amdgcn_wmma_f32_16x16x32_f16(false,(a),false,(b),(short)0,(c),false,false)
#define AFRAG(lo,hi) __builtin_shufflevector((lo),(hi),0,1,2,3,4,5,6,7,8,9,10,11,12,13,14,15)

// K-offset (0..31) of fragment half-slot j (0..7) for a lane in half `hf`
// (CDNA5 16-bit A/B layout: VGPR0..3 = K 0..7 / 8..15, VGPR4..7 = K 16..23 / 24..31).
__device__ __forceinline__ int akrow(int j, int hf) {
  return ((j >> 2) << 4) + ((j & 3) << 1) + (hf << 3);
}

// ---------------------------------------------------------------- utilities
__global__ void fill_zero_kernel(float* __restrict__ p, long n) {
  long t = (long)blockIdx.x * blockDim.x + threadIdx.x;
  if (t < n) p[t] = 0.f;
}

__global__ void dinv_kernel(const float* __restrict__ deg, float* __restrict__ dinv, int Nn) {
  int i = blockIdx.x * blockDim.x + threadIdx.x;
  if (i < Nn) {
    float d = deg[i];
    dinv[i] = d > 0.f ? __frsqrt_rn(d) : 0.f;
  }
}

__global__ void norm_kernel(const int* __restrict__ row, const int* __restrict__ col,
                            const float* __restrict__ ew, const float* __restrict__ dinv,
                            float* __restrict__ nrm, int E) {
  int e = blockIdx.x * blockDim.x + threadIdx.x;
  if (e < E) nrm[e] = -dinv[row[e]] * ew[e] * dinv[col[e]];
}

// out[col[e], :] += x[row[e], :] * nrm[e]   (4 channels per thread, coalesced)
__global__ void spmm_scatter_kernel(const float* __restrict__ x, int ldx,
                                    const float* __restrict__ nrm,
                                    const int* __restrict__ row, const int* __restrict__ col,
                                    float* __restrict__ out, int F, int E) {
  long t = (long)blockIdx.x * blockDim.x + threadIdx.x;
  int f4 = F >> 2;
  long total = (long)E * f4;
  if (t >= total) return;
  int e = (int)(t / f4);
  int f = (int)(t % f4) << 2;
  float w = nrm[e];
  const float* xp = x + (long)row[e] * ldx + f;
  float* op = out + (long)col[e] * F + f;
  atomicAdd(op + 0, xp[0] * w);
  atomicAdd(op + 1, xp[1] * w);
  atomicAdd(op + 2, xp[2] * w);
  atomicAdd(op + 3, xp[3] * w);
}

// -------- pack f32 weight (K x Nw, row-major) into f16 B-fragment order:
// out[((ntile*(K/32)+kc)*32 + lane)*16 + t]  -> one contiguous v16h per lane/WMMA
__global__ void pack_bfrag_kernel(const float* __restrict__ W, int K, int Kvalid, int Nw,
                                  _Float16* __restrict__ out) {
  int idx = blockIdx.x * blockDim.x + threadIdx.x;
  int KC = K >> 5;
  int total = (Nw >> 4) * KC * 32;
  if (idx >= total) return;
  int lane = idx & 31;
  int kc = (idx >> 5) % KC;
  int ntile = (idx >> 5) / KC;
  int m = lane & 15, hf = lane >> 4;
  int n = ntile * 16 + m;
  v16h vv;
  #pragma unroll
  for (int j = 0; j < 8; ++j) {
    int k = kc * 32 + akrow(j, hf);
    vv[2 * j]     = (_Float16)((k     < Kvalid) ? W[(long)k * Nw + n]       : 0.f);
    vv[2 * j + 1] = (_Float16)((k + 1 < Kvalid) ? W[(long)(k + 1) * Nw + n] : 0.f);
  }
  *(v16h*)(out + (long)idx * 16) = vv;
}

// ----------------------------------------------- PAE edge MLP + cosine -> ew, deg
// Per wave: 16 edges. GEMM1 (16x10)@(10x128) K-padded to 32; BN+ReLU;
// GEMM2 (16x128)@(128x128); cosine of the two halves.
__global__ __launch_bounds__(64) void pae_edge_kernel(
    const float* __restrict__ eni,                      // (E, 20)
    const _Float16* __restrict__ pw1,                   // packed (8 nt, 1 kc)
    const float* __restrict__ b1v,
    const float* __restrict__ gma, const float* __restrict__ bta,
    const float* __restrict__ mu,  const float* __restrict__ vr,
    const _Float16* __restrict__ pw2,                   // packed (8 nt, 4 kc)
    const float* __restrict__ b2v,
    const int* __restrict__ row,
    float* __restrict__ ew, float* __restrict__ deg, int E) {
  __shared__ __align__(16) _Float16 sX[2][2][16][40];   // [wave][half][row][k pad]
  __shared__ __align__(16) _Float16 sH[2][16][136];     // hidden tile (padded)
  __shared__ __align__(16) float    sOut[2][2][16][132];
  int lane = threadIdx.x & 31, wv = threadIdx.x >> 5;
  int tile = blockIdx.x * 2 + wv;
  int e0 = tile * 16;
  int m = lane & 15, hf = lane >> 4;

  {                                                     // stage inputs: row m, half hf
    int er = e0 + m;
    float msk = (er < E) ? 1.f : 0.f;
    long ce = (er < E) ? er : (long)(E - 1);
    const float* p = eni + ce * 20 + hf * 10;
    float xv[10];
    #pragma unroll
    for (int i = 0; i < 5; ++i) {
      float2 v = ((const float2*)p)[i];
      xv[2 * i] = v.x * msk; xv[2 * i + 1] = v.y * msk;
    }
    v8h* dst = (v8h*)&sX[wv][hf][m][0];
    #pragma unroll
    for (int i = 0; i < 4; ++i) {
      v8h t;
      #pragma unroll
      for (int q = 0; q < 8; ++q) {
        int k = 8 * i + q;
        t[q] = (_Float16)((k < 10) ? xv[k] : 0.f);
      }
      dst[i] = t;
    }
  }
  __syncthreads();

  for (int hx = 0; hx < 2; ++hx) {
    v8h lo = *(const v8h*)&sX[wv][hx][m][hf * 8];
    v8h hi = *(const v8h*)&sX[wv][hx][m][16 + hf * 8];
    v16h a = AFRAG(lo, hi);
    #pragma unroll
    for (int nt = 0; nt < 8; ++nt) {
      v16h b = *(const v16h*)(pw1 + ((long)nt * 32 + lane) * 16);
      v8f c = {};
      c = WMMA(a, b, c);
      int n = nt * 16 + m;
      float sc = gma[n] * __frsqrt_rn(vr[n] + 1e-5f);
      float sh = bta[n] - mu[n] * sc;
      float bb = b1v[n];
      #pragma unroll
      for (int r = 0; r < 8; ++r) {
        float x = fmaxf(c[r] + bb, 0.f) * sc + sh;      // relu then bn_eval
        sH[wv][r + 8 * hf][n] = (_Float16)x;
      }
    }
    __syncthreads();
    #pragma unroll
    for (int nt = 0; nt < 8; ++nt) {
      v8f c = {};
      #pragma unroll
      for (int kc = 0; kc < 4; ++kc) {
        v8h l2 = *(const v8h*)&sH[wv][m][kc * 32 + hf * 8];
        v8h h2 = *(const v8h*)&sH[wv][m][kc * 32 + 16 + hf * 8];
        v16h af = AFRAG(l2, h2);
        v16h bf = *(const v16h*)(pw2 + (((long)nt * 4 + kc) * 32 + lane) * 16);
        c = WMMA(af, bf, c);
      }
      int n = nt * 16 + m;
      float bb = b2v[n];
      #pragma unroll
      for (int r = 0; r < 8; ++r)
        sOut[wv][hx][r + 8 * hf][n] = c[r] + bb;
    }
    __syncthreads();
  }

  if (lane < 16) {                                      // cosine -> ew, degree
    int e = e0 + lane;
    if (e < E) {
      float dot = 0.f, na = 0.f, nb = 0.f;
      for (int k = 0; k < 128; ++k) {
        float x = sOut[wv][0][lane][k];
        float y = sOut[wv][1][lane][k];
        dot += x * y; na += x * x; nb += y * y;
      }
      float n1 = fmaxf(sqrtf(na), 1e-8f);
      float n2 = fmaxf(sqrtf(nb), 1e-8f);
      float w = 0.5f * (dot / (n1 * n2) + 1.f);
      ew[e] = w;
      atomicAdd(&deg[row[e]], w);
    }
  }
}

// ----------------- Cheb combine: out = relu(Tx0@W0 + T1@W1 + (2*T2 - Tx0)@W2)
// K outermost; 6 N-tile accumulators held in registers (18 WMMA per K-chunk).
__global__ __launch_bounds__(64) void cheb_combine_kernel(
    const float* __restrict__ x0, int ldx0,
    const float* __restrict__ t1, const float* __restrict__ t2, int ldt,
    const _Float16* __restrict__ pw, int KC,            // per-matrix stride 3072*KC halfs
    float* __restrict__ out, int ldo, int Nn) {
  __shared__ __align__(16) _Float16 sA[2][3][16][40];
  int lane = threadIdx.x & 31, wv = threadIdx.x >> 5;
  int tile = blockIdx.x * 2 + wv;
  int m = lane & 15, hf = lane >> 4;
  int mrow = tile * 16 + m;
  float msk = (mrow < Nn) ? 1.f : 0.f;
  long crow = (mrow < Nn) ? mrow : (long)(Nn - 1);
  long ps = (long)3072 * KC;
  v8f acc[6];
  #pragma unroll
  for (int i = 0; i < 6; ++i) acc[i] = (v8f){0, 0, 0, 0, 0, 0, 0, 0};

  for (int kc = 0; kc < KC; ++kc) {
    {                                                   // stage Tx0, T1, 2*T2-Tx0
      const float4* p0 = (const float4*)(x0 + crow * ldx0 + kc * 32 + hf * 16);
      const float4* p1 = (const float4*)(t1 + crow * ldt + kc * 32 + hf * 16);
      const float4* p2 = (const float4*)(t2 + crow * ldt + kc * 32 + hf * 16);
      if (kc + 1 < KC) {
        __builtin_prefetch(x0 + crow * ldx0 + (kc + 1) * 32 + hf * 16, 0, 1);
        __builtin_prefetch(t1 + crow * ldt + (kc + 1) * 32 + hf * 16, 0, 1);
        __builtin_prefetch(t2 + crow * ldt + (kc + 1) * 32 + hf * 16, 0, 1);
      }
      v8h o0[2], o1[2], o2[2];
      #pragma unroll
      for (int i = 0; i < 4; ++i) {
        float4 a4 = p0[i], b4 = p1[i], c4 = p2[i];
        int q = (i & 1) * 4;
        v8h* d0 = &o0[i >> 1]; v8h* d1 = &o1[i >> 1]; v8h* d2 = &o2[i >> 1];
        (*d0)[q + 0] = (_Float16)(a4.x * msk); (*d0)[q + 1] = (_Float16)(a4.y * msk);
        (*d0)[q + 2] = (_Float16)(a4.z * msk); (*d0)[q + 3] = (_Float16)(a4.w * msk);
        (*d1)[q + 0] = (_Float16)(b4.x * msk); (*d1)[q + 1] = (_Float16)(b4.y * msk);
        (*d1)[q + 2] = (_Float16)(b4.z * msk); (*d1)[q + 3] = (_Float16)(b4.w * msk);
        (*d2)[q + 0] = (_Float16)((2.f * c4.x - a4.x) * msk);
        (*d2)[q + 1] = (_Float16)((2.f * c4.y - a4.y) * msk);
        (*d2)[q + 2] = (_Float16)((2.f * c4.z - a4.z) * msk);
        (*d2)[q + 3] = (_Float16)((2.f * c4.w - a4.w) * msk);
      }
      *(v8h*)&sA[wv][0][m][hf * 16] = o0[0]; *(v8h*)&sA[wv][0][m][hf * 16 + 8] = o0[1];
      *(v8h*)&sA[wv][1][m][hf * 16] = o1[0]; *(v8h*)&sA[wv][1][m][hf * 16 + 8] = o1[1];
      *(v8h*)&sA[wv][2][m][hf * 16] = o2[0]; *(v8h*)&sA[wv][2][m][hf * 16 + 8] = o2[1];
    }
    __syncthreads();
    v16h a0 = AFRAG(*(const v8h*)&sA[wv][0][m][hf * 8], *(const v8h*)&sA[wv][0][m][16 + hf * 8]);
    v16h a1 = AFRAG(*(const v8h*)&sA[wv][1][m][hf * 8], *(const v8h*)&sA[wv][1][m][16 + hf * 8]);
    v16h a2 = AFRAG(*(const v8h*)&sA[wv][2][m][hf * 8], *(const v8h*)&sA[wv][2][m][16 + hf * 8]);
    #pragma unroll
    for (int nt = 0; nt < 6; ++nt) {
      long bi = ((long)(nt * KC + kc) * 32 + lane) * 16;
      v16h b0 = *(const v16h*)(pw + bi);
      v16h b1 = *(const v16h*)(pw + ps + bi);
      v16h b2 = *(const v16h*)(pw + 2 * ps + bi);
      acc[nt] = WMMA(a0, b0, acc[nt]);
      acc[nt] = WMMA(a1, b1, acc[nt]);
      acc[nt] = WMMA(a2, b2, acc[nt]);
    }
    __syncthreads();
  }
  #pragma unroll
  for (int nt = 0; nt < 6; ++nt) {
    int n = nt * 16 + m;
    #pragma unroll
    for (int r = 0; r < 8; ++r) {
      int mm = tile * 16 + r + 8 * hf;
      if (mm < Nn) out[(long)mm * ldo + n] = fmaxf(acc[nt][r], 0.f);
    }
  }
}

// ----------------- classifier: relu(jk@w1+b1) -> bn -> @w2+b2 -> log_softmax
__global__ __launch_bounds__(64) void cls_kernel(
    const float* __restrict__ jk,                       // (N, 384)
    const _Float16* __restrict__ pw1,                   // packed (16 nt, 12 kc)
    const float* __restrict__ b1v,
    const float* __restrict__ gma, const float* __restrict__ bta,
    const float* __restrict__ mu,  const float* __restrict__ vr,
    const float* __restrict__ w2, const float* __restrict__ b2v,
    float* __restrict__ outlog, int Nn) {
  __shared__ __align__(16) _Float16 sA[2][16][392];
  __shared__ float sZ[2][16][260];
  __shared__ float sW2[512];
  __shared__ float sL[2][16][2];
  int lane = threadIdx.x & 31, wv = threadIdx.x >> 5;
  int tile = blockIdx.x * 2 + wv;
  int m = lane & 15, hf = lane >> 4;
  int mrow = tile * 16 + m;
  float msk = (mrow < Nn) ? 1.f : 0.f;
  long crow = (mrow < Nn) ? mrow : (long)(Nn - 1);

  {                                                     // cls_w2 (256x2) -> LDS
    int t = threadIdx.x;
    float4 a4 = ((const float4*)w2)[t * 2];
    float4 b4 = ((const float4*)w2)[t * 2 + 1];
    ((float4*)sW2)[t * 2] = a4;
    ((float4*)sW2)[t * 2 + 1] = b4;
  }
  #pragma unroll
  for (int kc = 0; kc < 12; ++kc) {                     // stage jk tile (16x384) f16
    const float4* p = (const float4*)(jk + crow * 384 + kc * 32 + hf * 16);
    v8h o[2];
    #pragma unroll
    for (int i = 0; i < 4; ++i) {
      float4 a4 = p[i];
      int q = (i & 1) * 4;
      v8h* d = &o[i >> 1];
      (*d)[q + 0] = (_Float16)(a4.x * msk); (*d)[q + 1] = (_Float16)(a4.y * msk);
      (*d)[q + 2] = (_Float16)(a4.z * msk); (*d)[q + 3] = (_Float16)(a4.w * msk);
    }
    *(v8h*)&sA[wv][m][kc * 32 + hf * 16] = o[0];
    *(v8h*)&sA[wv][m][kc * 32 + hf * 16 + 8] = o[1];
  }
  __syncthreads();
  #pragma unroll 4
  for (int nt = 0; nt < 16; ++nt) {
    v8f c = {};
    #pragma unroll
    for (int kc = 0; kc < 12; ++kc) {
      v16h a = AFRAG(*(const v8h*)&sA[wv][m][kc * 32 + hf * 8],
                     *(const v8h*)&sA[wv][m][kc * 32 + 16 + hf * 8]);
      v16h b = *(const v16h*)(pw1 + ((long)(nt * 12 + kc) * 32 + lane) * 16);
      c = WMMA(a, b, c);
    }
    int n = nt * 16 + m;
    float sc = gma[n] * __frsqrt_rn(vr[n] + 1e-5f);
    float sh = bta[n] - mu[n] * sc;
    float bb = b1v[n];
    #pragma unroll
    for (int r = 0; r < 8; ++r) {
      float z = fmaxf(c[r] + bb, 0.f);                  // relu first
      sZ[wv][r + 8 * hf][n] = z * sc + sh;              // then bn_eval
    }
  }
  __syncthreads();
  int cc = hf;                                          // lanes 0-15: class 0; 16-31: class 1
  float accv = b2v[cc];
  for (int k = 0; k < 256; ++k) accv += sZ[wv][m][k] * sW2[k * 2 + cc];
  sL[wv][m][cc] = accv;
  __syncthreads();
  if (mrow < Nn) {
    float l0 = sL[wv][m][0], l1 = sL[wv][m][1];
    float mx = fmaxf(l0, l1);
    float lse = mx + logf(__expf(l0 - mx) + __expf(l1 - mx));
    outlog[(long)mrow * 2 + cc] = sL[wv][m][cc] - lse;
  }
}

// ------------------------------------------------------------------ launcher
extern "C" void kernel_launch(void* const* d_in, const int* in_sizes, int n_in,
                              void* d_out, int out_size, void* d_ws, size_t ws_size,
                              hipStream_t stream) {
  const float* features = (const float*)d_in[0];
  const int*   eidx     = (const int*)d_in[1];
  const float* eni      = (const float*)d_in[2];
  const float* pae_w1   = (const float*)d_in[3];
  const float* pae_b1   = (const float*)d_in[4];
  const float* pae_g    = (const float*)d_in[5];
  const float* pae_be   = (const float*)d_in[6];
  const float* pae_m    = (const float*)d_in[7];
  const float* pae_v    = (const float*)d_in[8];
  const float* pae_w2   = (const float*)d_in[9];
  const float* pae_b2   = (const float*)d_in[10];
  const float* cheb_w[4] = {(const float*)d_in[11], (const float*)d_in[12],
                            (const float*)d_in[13], (const float*)d_in[14]};
  const float* cls_w1 = (const float*)d_in[15];
  const float* cls_b1 = (const float*)d_in[16];
  const float* cls_g  = (const float*)d_in[17];
  const float* cls_be = (const float*)d_in[18];
  const float* cls_m  = (const float*)d_in[19];
  const float* cls_v  = (const float*)d_in[20];
  const float* cls_w2 = (const float*)d_in[21];
  const float* cls_b2 = (const float*)d_in[22];

  const int N = in_sizes[0] / 256;     // 50000
  const int E = in_sizes[1] / 2;       // 800000
  const int* row = eidx;
  const int* col = eidx + E;

  float* jk     = (float*)d_out;                 // (N, 384)
  float* logits = jk + (long)N * 384;            // (N, 2)

  float* ws   = (float*)d_ws;
  float* ew   = ws;                              // E
  float* deg  = ew + E;                          // N
  float* dinv = deg + N;                         // N
  float* nrm  = dinv + N;                        // E
  float* T1   = nrm + E;                         // N*256
  float* T2   = T1 + (long)N * 256;              // N*256
  _Float16* pw1    = (_Float16*)(T2 + (long)N * 256);   // 4096
  _Float16* pw2    = pw1 + 4096;                        // 16384
  _Float16* pch[4];
  pch[0] = pw2 + 16384;                                 // 3*24576
  pch[1] = pch[0] + 73728;                              // 3*9216
  pch[2] = pch[1] + 27648;
  pch[3] = pch[2] + 27648;
  _Float16* pcls = pch[3] + 27648;                      // 98304

  // 0) pack all weights into f16 B-fragment order (on-stream, every call)
  {
    int t1n = (128 / 16) * (32 / 32) * 32;
    pack_bfrag_kernel<<<(t1n + 255) / 256, 256, 0, stream>>>(pae_w1, 32, 10, 128, pw1);
    int t2n = (128 / 16) * (128 / 32) * 32;
    pack_bfrag_kernel<<<(t2n + 255) / 256, 256, 0, stream>>>(pae_w2, 128, 128, 128, pw2);
    for (int L = 0; L < 4; ++L) {
      int Fin = (L == 0) ? 256 : 96;
      int KC = Fin / 32;
      int tn = 6 * KC * 32;
      for (int i = 0; i < 3; ++i)
        pack_bfrag_kernel<<<(tn + 255) / 256, 256, 0, stream>>>(
            cheb_w[L] + (long)i * Fin * 96, Fin, Fin, 96, pch[L] + (long)i * 3072 * KC);
    }
    int tcn = 16 * 12 * 32;
    pack_bfrag_kernel<<<(tcn + 255) / 256, 256, 0, stream>>>(cls_w1, 384, 384, 256, pcls);
  }

  const int etiles = (E + 15) / 16;
  const int rtiles = (N + 15) / 16;

  // 1) edge weights via PAE + degree
  fill_zero_kernel<<<(N + 255) / 256, 256, 0, stream>>>(deg, N);
  pae_edge_kernel<<<(etiles + 1) / 2, 64, 0, stream>>>(
      eni, pw1, pae_b1, pae_g, pae_be, pae_m, pae_v, pw2, pae_b2, row, ew, deg, E);
  dinv_kernel<<<(N + 255) / 256, 256, 0, stream>>>(deg, dinv, N);
  norm_kernel<<<(E + 255) / 256, 256, 0, stream>>>(row, col, ew, dinv, nrm, E);

  // 2) four ChebConv layers, JK-concat written straight into d_out
  for (int L = 0; L < 4; ++L) {
    const int   Fin  = (L == 0) ? 256 : 96;
    const float* x0  = (L == 0) ? features : (jk + 96 * (L - 1));
    const int   ldx0 = (L == 0) ? 256 : 384;
    long nf = (long)N * Fin;
    long sp = (long)E * (Fin >> 2);
    fill_zero_kernel<<<(nf + 255) / 256, 256, 0, stream>>>(T1, nf);
    spmm_scatter_kernel<<<(sp + 255) / 256, 256, 0, stream>>>(
        x0, ldx0, nrm, row, col, T1, Fin, E);
    fill_zero_kernel<<<(nf + 255) / 256, 256, 0, stream>>>(T2, nf);
    spmm_scatter_kernel<<<(sp + 255) / 256, 256, 0, stream>>>(
        T1, Fin, nrm, row, col, T2, Fin, E);
    cheb_combine_kernel<<<(rtiles + 1) / 2, 64, 0, stream>>>(
        x0, ldx0, T1, T2, Fin, pch[L], Fin / 32, jk + 96 * L, 384, N);
  }

  // 3) classifier + log-softmax
  cls_kernel<<<(rtiles + 1) / 2, 64, 0, stream>>>(
      jk, pcls, cls_b1, cls_g, cls_be, cls_m, cls_v, cls_w2, cls_b2, logits, N);
}
